// NonLocalBlock3D_231928234272
// MI455X (gfx1250) — 8-bit matrix operand order repaired and checked by reading, compiled, not run on hardware
//
#include <hip/hip_runtime.h>

#define BATCH 4
#define CCH   128
#define NPOS  4096
#define DIN   64
#define QSCALE 0.18033688011112042f
#define PSTR 72

typedef __attribute__((ext_vector_type(8))) int   v8i;
typedef __attribute__((ext_vector_type(8))) float v8f;

union Frag8 { uint2 u2[4]; uint4 u4[2]; v8i v; };

#if __has_builtin(__builtin_amdgcn_cvt_pk_fp8_f32)
#define HAVE_CVT_FP8 1
#else
#define HAVE_CVT_FP8 0
#endif

#if __has_builtin(__builtin_amdgcn_permlane16)
#define HAVE_PERMLANE16 1
#else
#define HAVE_PERMLANE16 0
#endif

__device__ __forceinline__ float fast_exp2(float x) {
#if __has_builtin(__builtin_amdgcn_exp2f)
  return __builtin_amdgcn_exp2f(x);
#else
  return __builtin_exp2f(x);
#endif
}

__device__ __forceinline__ unsigned char f2fp8(float f) {
#if HAVE_CVT_FP8
  return (unsigned char)(__builtin_amdgcn_cvt_pk_fp8_f32(f, 0.0f, 0, false) & 0xFF);
#else
  float cf = fminf(fmaxf(f, -448.0f), 448.0f);
  unsigned u = __float_as_uint(cf);
  unsigned sgn = (u >> 24) & 0x80u;
  int ee = (int)((u >> 23) & 0xFF) - 127 + 7;
  unsigned m = (u >> 20) & 7u;
  unsigned out = (ee <= 0) ? 0u : (((unsigned)ee << 3) | m);
  return (unsigned char)(sgn | out);
#endif
}

__device__ __forceinline__ unsigned int pack4_fp8(float a, float b, float c, float d) {
#if HAVE_CVT_FP8
  unsigned int w = (unsigned int)__builtin_amdgcn_cvt_pk_fp8_f32(a, b, 0, false);
  w = (unsigned int)__builtin_amdgcn_cvt_pk_fp8_f32(c, d, (int)w, true);
  return w;
#else
  return (unsigned)f2fp8(a) | ((unsigned)f2fp8(b) << 8) |
         ((unsigned)f2fp8(c) << 16) | ((unsigned)f2fp8(d) << 24);
#endif
}

#if HAVE_PERMLANE16
__device__ __forceinline__ float permx(float v, int s0, int s1) {
  return __uint_as_float((unsigned)__builtin_amdgcn_permlane16(
      (int)__float_as_uint(v), (int)__float_as_uint(v), s0, s1, false, false));
}
#endif

__device__ __forceinline__ void rowmax16(float m[8]) {
#if HAVE_PERMLANE16
  const int sel[4][2] = {
      {(int)0xFEDCBA98, (int)0x76543210},
      {(int)0x32107654, (int)0xBA98FEDC},
      {(int)0x54761032, (int)0xDCFE98BA},
      {(int)0x67452301, (int)0xEFCDAB89}};
#pragma unroll
  for (int st = 0; st < 4; ++st) {
    float t[8];
#pragma unroll
    for (int r = 0; r < 8; ++r) t[r] = permx(m[r], sel[st][0], sel[st][1]);
#pragma unroll
    for (int r = 0; r < 8; ++r) m[r] = fmaxf(m[r], t[r]);
  }
#else
#pragma unroll
  for (int off = 8; off >= 1; off >>= 1) {
    float t[8];
#pragma unroll
    for (int r = 0; r < 8; ++r) t[r] = __shfl_xor(m[r], off, 16);
#pragma unroll
    for (int r = 0; r < 8; ++r) m[r] = fmaxf(m[r], t[r]);
  }
#endif
}

__device__ __forceinline__ void load_fragA8(Frag8& f, const unsigned char* base, int g) {
  f.u2[0] = *(const uint2*)(base + 8 * g);
  f.u2[1] = *(const uint2*)(base + 16 + 8 * g);
  f.u2[2] = *(const uint2*)(base + 32 + 8 * g);
  f.u2[3] = *(const uint2*)(base + 48 + 8 * g);
}
__device__ __forceinline__ void load_fragB8(Frag8& f, const unsigned char* base, int g) {
  f.u2[0] = *(const uint2*)(base + 8 * g);
  f.u2[1] = *(const uint2*)(base + 16 + 8 * g);
  f.u2[2] = *(const uint2*)(base + 32 + 8 * g);
  f.u2[3] = *(const uint2*)(base + 48 + 8 * g);
}

__device__ __forceinline__ v8f wmma_fp8(v8i a, v8i b, v8f c) {
  return __builtin_amdgcn_wmma_f32_16x16x64_fp8_fp8(a, b, (short)0, c, false, false);
}

__global__ __launch_bounds__(256) void qkv_proj_kernel(
    const float* __restrict__ x,
    const float* __restrict__ Wq, const float* __restrict__ bq,
    const float* __restrict__ Wk, const float* __restrict__ bk,
    const float* __restrict__ Wv, const float* __restrict__ bv,
    unsigned char* __restrict__ Qb, unsigned char* __restrict__ Kb,
    unsigned char* __restrict__ Vb)
{
  __shared__ float xs[CCH][64];
  const int tid = threadIdx.x;
  const int blk = blockIdx.x;
  const int b   = blk >> 6;
  const int n0  = (blk & 63) << 6;

  const float* xb = x + ((size_t)b * CCH) * NPOS + n0;
  for (int i = tid; i < CCH * 64; i += 256) {
    int c = i >> 6, p = i & 63;
    xs[c][p] = xb[(size_t)c * NPOS + p];
  }
  __syncthreads();

  const int oid = tid & 31;
  const int pg  = tid >> 5;

  for (int ob = 0; ob < 8; ++ob) {
    const int ro = ob * 32 + oid;
    const float* wrow; float bias; float osc;
    if (ro < 64)        { wrow = Wq + ro * CCH;         bias = bq[ro];        osc = QSCALE; }
    else if (ro < 128)  { wrow = Wk + (ro - 64) * CCH;  bias = bk[ro - 64];   osc = 1.0f;   }
    else                { wrow = Wv + (ro - 128) * CCH; bias = bv[ro - 128];  osc = 1.0f;   }

    float acc[8];
#pragma unroll
    for (int p = 0; p < 8; ++p) acc[p] = bias;
#pragma unroll 4
    for (int c = 0; c < CCH; ++c) {
      float w = wrow[c];
#pragma unroll
      for (int p = 0; p < 8; ++p) acc[p] = fmaf(w, xs[c][pg * 8 + p], acc[p]);
    }

    if (ro < 128) {
      unsigned char* dst = (ro < 64)
          ? (Qb + ((size_t)(b * NPOS + n0 + pg * 8)) * DIN + ro)
          : (Kb + ((size_t)(b * NPOS + n0 + pg * 8)) * DIN + (ro - 64));
#pragma unroll
      for (int p = 0; p < 8; ++p) dst[(size_t)p * DIN] = f2fp8(acc[p] * osc);
    } else {
      unsigned char* vdst = Vb + ((size_t)(b * CCH + (ro - 128))) * NPOS + n0;
#pragma unroll
      for (int p = 0; p < 8; ++p) {
        int z = pg * 8 + p;
        vdst[(z & 15) * 4 + (z >> 4)] = f2fp8(acc[p]);
      }
    }
  }
}

__global__ __launch_bounds__(128) void attn_kernel(
    const float* __restrict__ x, const float* __restrict__ gammap,
    const unsigned char* __restrict__ Qb, const unsigned char* __restrict__ Kb,
    const unsigned char* __restrict__ Vb, float* __restrict__ out)
{
  __shared__ unsigned char Plds[4][16 * PSTR];
  __shared__ float Obuf[4][64][32];
  __shared__ float Smax[4][8][32];
  __shared__ float Ssum[4][8][32];

  const int tid  = threadIdx.x;
  const int lane = tid & 31;
  const int widx = tid >> 5;
  const int b    = blockIdx.x >> 8;
  const int n0   = (blockIdx.x & 255) << 4;
  const int g    = lane >> 4;
  const int mr   = lane & 15;

  const unsigned char* Kbb = Kb + ((size_t)b * NPOS) * DIN;
  const unsigned char* Vbb = Vb + ((size_t)b * CCH) * NPOS;

  Frag8 qf;
  load_fragA8(qf, Qb + ((size_t)(b * NPOS + n0 + mr)) * DIN, g);

  Frag8 ones;
#pragma unroll
  for (int i = 0; i < 4; ++i) { ones.u4[0][i] = 0x38383838u; ones.u4[1][i] = 0x38383838u; }

  v8f Oacc[8], Osum;
#pragma unroll
  for (int ct = 0; ct < 8; ++ct)
#pragma unroll
    for (int r = 0; r < 8; ++r) Oacc[ct][r] = 0.0f;
#pragma unroll
  for (int r = 0; r < 8; ++r) Osum[r] = 0.0f;

  float runmax[8];
#pragma unroll
  for (int r = 0; r < 8; ++r) runmax[r] = -1.0e30f;

  unsigned char* Pw = &Plds[widx][0];

  for (int it = 0; it < 16; ++it) {
    const int m0 = (it << 8) + (widx << 6);

    if (it < 15)
      __builtin_prefetch(Kbb + ((size_t)(m0 + 256 + mr)) * DIN, 0, 3);

    Frag8 kf[4];
#pragma unroll
    for (int t = 0; t < 4; ++t)
      load_fragB8(kf[t], Kbb + ((size_t)(m0 + t * 16 + mr)) * DIN, g);

    v8f s[4];
#pragma unroll
    for (int t = 0; t < 4; ++t) {
      v8f acc;
#pragma unroll
      for (int r = 0; r < 8; ++r) acc[r] = 0.0f;
      s[t] = wmma_fp8(qf.v, kf[t].v, acc);
    }

    float m[8];
#pragma unroll
    for (int r = 0; r < 8; ++r)
      m[r] = fmaxf(fmaxf(s[0][r], s[1][r]), fmaxf(s[2][r], s[3][r]));
    rowmax16(m);

    float corr[8];
#pragma unroll
    for (int r = 0; r < 8; ++r) {
      float nm = fmaxf(runmax[r], m[r]);
      corr[r] = fast_exp2(runmax[r] - nm);
      runmax[r] = nm;
#pragma unroll
      for (int t = 0; t < 4; ++t) s[t][r] = fast_exp2(s[t][r] - nm);
    }
    v8f cvec;
#pragma unroll
    for (int r = 0; r < 8; ++r) cvec[r] = corr[r];
#pragma unroll
    for (int ct = 0; ct < 8; ++ct) Oacc[ct] *= cvec;
    Osum *= cvec;

    Frag8 vf[2];
    load_fragB8(vf[0], Vbb + ((size_t)mr) * NPOS + m0, g);

#pragma unroll
    for (int r = 0; r < 8; ++r)
      *(unsigned int*)(Pw + (8 * g + r) * PSTR + 4 * mr) =
          pack4_fp8(s[0][r], s[1][r], s[2][r], s[3][r]);
    asm volatile("s_wait_dscnt 0" ::: "memory");

    Frag8 pf;
    load_fragA8(pf, Pw + mr * PSTR, g);

    Osum = wmma_fp8(pf.v, ones.v, Osum);

#pragma unroll
    for (int ct = 0; ct < 8; ++ct) {
      const int cur = ct & 1, nxt = cur ^ 1;
      if (ct < 7)
        load_fragB8(vf[nxt], Vbb + ((size_t)((ct + 1) * 16 + mr)) * NPOS + m0, g);
      Oacc[ct] = wmma_fp8(pf.v, vf[cur].v, Oacc[ct]);
    }
  }

#pragma unroll
  for (int ct = 0; ct < 8; ++ct)
#pragma unroll
    for (int r = 0; r < 8; ++r) Obuf[widx][ct * 8 + r][lane] = Oacc[ct][r];
#pragma unroll
  for (int r = 0; r < 8; ++r) {
    Smax[widx][r][lane] = runmax[r];
    Ssum[widx][r][lane] = Osum[r];
  }
  __syncthreads();

  if (widx == 0) {
    float wsc[4][8], L[8];
#pragma unroll
    for (int r = 0; r < 8; ++r) {
      float mm = Smax[0][r][lane];
#pragma unroll
      for (int w = 1; w < 4; ++w) mm = fmaxf(mm, Smax[w][r][lane]);
      float l = 0.0f;
#pragma unroll
      for (int w = 0; w < 4; ++w) {
        wsc[w][r] = fast_exp2(Smax[w][r][lane] - mm);
        l += Ssum[w][r][lane] * wsc[w][r];
      }
      L[r] = 1.0f / l;
    }
    const float gamma = gammap[0];
#pragma unroll
    for (int ct = 0; ct < 8; ++ct) {
      const int c = ct * 16 + mr;
      const size_t base = ((size_t)(b * CCH + c)) * NPOS + n0 + 8 * g;
      float4 x0 = *(const float4*)(x + base);
      float4 x1 = *(const float4*)(x + base + 4);
      float o[8];
#pragma unroll
      for (int r = 0; r < 8; ++r) {
        float v = 0.0f;
#pragma unroll
        for (int w = 0; w < 4; ++w) v += Obuf[w][ct * 8 + r][lane] * wsc[w][r];
        o[r] = v * L[r];
      }
      float4 o0, o1;
      o0.x = x0.x + gamma * o[0]; o0.y = x0.y + gamma * o[1];
      o0.z = x0.z + gamma * o[2]; o0.w = x0.w + gamma * o[3];
      o1.x = x1.x + gamma * o[4]; o1.y = x1.y + gamma * o[5];
      o1.z = x1.z + gamma * o[6]; o1.w = x1.w + gamma * o[7];
      *(float4*)(out + base)     = o0;
      *(float4*)(out + base + 4) = o1;
    }
  }
}

extern "C" void kernel_launch(void* const* d_in, const int* in_sizes, int n_in,
                              void* d_out, int out_size, void* d_ws, size_t ws_size,
                              hipStream_t stream) {
  const float* x     = (const float*)d_in[0];
  const float* Wq    = (const float*)d_in[1];
  const float* bq    = (const float*)d_in[2];
  const float* Wk    = (const float*)d_in[3];
  const float* bk    = (const float*)d_in[4];
  const float* Wv    = (const float*)d_in[5];
  const float* bv    = (const float*)d_in[6];
  const float* gamma = (const float*)d_in[7];
  float* out = (float*)d_out;

  unsigned char* Qb = (unsigned char*)d_ws;
  unsigned char* Kb = Qb + (size_t)BATCH * NPOS * DIN;
  unsigned char* Vb = Kb + (size_t)BATCH * NPOS * DIN;

  qkv_proj_kernel<<<BATCH * (NPOS / 64), 256, 0, stream>>>(
      x, Wq, bq, Wk, bk, Wv, bv, Qb, Kb, Vb);

  attn_kernel<<<BATCH * (NPOS / 16), 128, 0, stream>>>(
      x, gamma, Qb, Kb, Vb, out);
}
